// CausalSelfAttention_25855703122718
// MI455X (gfx1250) — matrix operand order repaired and checked by reading, not compiled, not run on hardware
//
#include <hip/hip_runtime.h>
#include <hip/hip_bf16.h>
#include <stdint.h>

// ---------------- problem constants ----------------
constexpr int TSEQ = 2048;
constexpr int DIM  = 2048;
constexpr int NH   = 16;
constexpr int HD   = 128;           // head dim
constexpr float EPS = 1.1920929e-07f;
constexpr float ATT_SCALE = 0.08838834764831845f;   // 1/sqrt(128)
constexpr float LN1E4_OVER_64 = 0.14391156831212787f; // ln(10000)/64

// ---------------- WMMA types ----------------
typedef __bf16 bf16_t;
typedef __attribute__((ext_vector_type(16))) __bf16 v16bf;
typedef __attribute__((ext_vector_type(8)))  float  v8f;
typedef __attribute__((ext_vector_type(4)))  unsigned int u32x4;
typedef __attribute__((ext_vector_type(4)))  int         i32x4;

struct U32x8 { u32x4 a, b; };

// ---- CDNA5 async global->LDS staging (ASYNCcnt path), with sync fallback ----
#if defined(__HIP_DEVICE_COMPILE__) && __has_builtin(__builtin_amdgcn_global_load_async_to_lds_b128)
#define ASYNC_LDS 1
#endif

typedef __attribute__((address_space(1))) i32x4 as1_i32x4;   // global
typedef __attribute__((address_space(3))) i32x4 as3_i32x4;   // LDS

__device__ inline void copy16B_g2l(const uint16_t* g, uint16_t* l) {
#ifdef ASYNC_LDS
  __builtin_amdgcn_global_load_async_to_lds_b128((as1_i32x4*)g, (as3_i32x4*)l, 0, 0);
#else
  *(u32x4*)l = *(const u32x4*)g;
#endif
}

__device__ inline void wait_stage(bool next_inflight) {
#ifdef ASYNC_LDS
  if (next_inflight) {
#if __has_builtin(__builtin_amdgcn_s_wait_asynccnt)
    __builtin_amdgcn_s_wait_asynccnt(4);   // 4 next-tile loads may remain in flight
#else
    asm volatile("s_wait_asynccnt 4" ::: "memory");
#endif
  } else {
#if __has_builtin(__builtin_amdgcn_s_wait_asynccnt)
    __builtin_amdgcn_s_wait_asynccnt(0);
#else
    asm volatile("s_wait_asynccnt 0" ::: "memory");
#endif
  }
#else
  (void)next_inflight;
#endif
}

__device__ inline uint16_t f2bf(float f) {
  uint32_t u = __builtin_bit_cast(uint32_t, f);
  uint32_t r = u + 0x7FFFu + ((u >> 16) & 1u);   // round to nearest even
  return (uint16_t)(r >> 16);
}

// A-fragment (16-bit, 16x32): lane holds two 8-element (16B) chunks
__device__ inline v16bf make_frag2(const uint16_t* c0, const uint16_t* c1) {
  U32x8 u;
  u.a = *(const u32x4*)c0;
  u.b = *(const u32x4*)c1;
  return __builtin_bit_cast(v16bf, u);
}
// B-fragment (16-bit, 32x16): lane holds two 8-element (16B) chunks at c and c+16
__device__ inline v16bf make_frag16(const uint16_t* c) {
  U32x8 u;
  u.a = *(const u32x4*)c;
  u.b = *(const u32x4*)(c + 16);
  return __builtin_bit_cast(v16bf, u);
}

__device__ inline v8f wmma_bf16(v16bf a, v16bf b, v8f c) {
  return __builtin_amdgcn_wmma_f32_16x16x32_bf16(
      /*neg_a=*/false, a, /*neg_b=*/false, b,
      /*c_mod=*/(short)0, c, /*reuse_a=*/false, /*reuse_b=*/false);
}

// ---------------- elementwise kernels ----------------
__global__ __launch_bounds__(256) void k_f32_to_bf16(const float* __restrict__ in,
                                                     uint16_t* __restrict__ out, int n) {
  int i = blockIdx.x * 256 + threadIdx.x;
  if (i < n) out[i] = f2bf(in[i]);
}

__global__ __launch_bounds__(256) void k_copy_f32(const float* __restrict__ in,
                                                  float* __restrict__ out, int n) {
  int i = blockIdx.x * 256 + threadIdx.x;
  if (i < n) out[i] = in[i];
}

// ---------------- bf16 GEMM: C[M,N] = A[M,K] * W[N,K]^T ----------------
// Tile 128x128, K-step 32, double-buffered LDS with async global->LDS staging.
// 8 waves/block; each wave: 32(M) x 64(N) via 2x4 WMMA fragments.
__global__ __launch_bounds__(256)
void k_gemm_bf16(const uint16_t* __restrict__ A,   // M x K, bf16 row-major
                 const uint16_t* __restrict__ W,   // N x K, bf16 row-major
                 float* __restrict__ C,            // M x N, f32 row-major
                 int M, int N, int K) {
  __shared__ uint16_t As[2][128][40];   // +8 pad keeps 16B alignment, dodges bank conflicts
  __shared__ uint16_t Bs[2][128][40];

  const int tid    = threadIdx.x;
  const int lane   = tid & 31;
  const int w      = tid >> 5;
  const int lanelo = lane & 15;
  const int half   = lane >> 4;
  const int wy = w >> 1;              // 0..3  -> 32-row strips
  const int wx = w & 1;               // 0..1  -> 64-col strips
  const int mbase = wy * 32;
  const int nbase = wx * 64;
  const int m0 = blockIdx.y * 128;
  const int n0 = blockIdx.x * 128;

  // per-thread staging footprint: 2x16B of A + 2x16B of B per tile
  const int id0  = tid;
  const int id1  = tid + 256;
  const int row0 = id0 >> 2, colc0 = (id0 & 3) * 8;
  const int row1 = id1 >> 2, colc1 = (id1 & 3) * 8;

  auto stage = [&](int buf, int kt) {
    copy16B_g2l(&A[(size_t)(m0 + row0) * K + kt + colc0], &As[buf][row0][colc0]);
    copy16B_g2l(&W[(size_t)(n0 + row0) * K + kt + colc0], &Bs[buf][row0][colc0]);
    copy16B_g2l(&A[(size_t)(m0 + row1) * K + kt + colc1], &As[buf][row1][colc1]);
    copy16B_g2l(&W[(size_t)(n0 + row1) * K + kt + colc1], &Bs[buf][row1][colc1]);
  };

  v8f acc[2][4];
#pragma unroll
  for (int mi = 0; mi < 2; ++mi)
#pragma unroll
    for (int ni = 0; ni < 4; ++ni) acc[mi][ni] = (v8f)(0.0f);

  stage(0, 0);
  int buf = 0;
  for (int kt = 0; kt < K; kt += 32) {
    const bool more = (kt + 32) < K;
    if (more) stage(buf ^ 1, kt + 32);   // prev reads of buf^1 fenced by last barrier
    wait_stage(more);                    // in-order ASYNCcnt: current tile has landed
    __syncthreads();

    v16bf af[2], bfv[4];
#pragma unroll
    for (int mi = 0; mi < 2; ++mi) {
      const uint16_t* ar = &As[buf][mbase + mi * 16 + lanelo][0];
      af[mi] = make_frag2(ar + half * 8, ar + 16 + half * 8);
    }
#pragma unroll
    for (int ni = 0; ni < 4; ++ni) {
      const uint16_t* br = &Bs[buf][nbase + ni * 16 + lanelo][0];
      bfv[ni] = make_frag16(br + half * 8);
    }
#pragma unroll
    for (int mi = 0; mi < 2; ++mi)
#pragma unroll
      for (int ni = 0; ni < 4; ++ni)
        acc[mi][ni] = wmma_bf16(af[mi], bfv[ni], acc[mi][ni]);

    __syncthreads();                     // all reads of buf done before it is re-staged
    buf ^= 1;
  }

  // D layout: VGPR r -> row r (lanes 0-15) / row 8+r (lanes 16-31); col = lane&15
#pragma unroll
  for (int mi = 0; mi < 2; ++mi)
#pragma unroll
    for (int ni = 0; ni < 4; ++ni)
#pragma unroll
      for (int r = 0; r < 8; ++r) {
        int row = m0 + mbase + mi * 16 + half * 8 + r;
        int col = n0 + nbase + ni * 16 + lanelo;
        C[(size_t)row * N + col] = acc[mi][ni][r];
      }
}

// ---------------- post: v-mix, rms_norm + rotary on q/k, bf16 emit ----------------
// grid (TSEQ, NH), block 128 (one thread per head-dim element)
__global__ __launch_bounds__(128)
void k_qkv_post(const float* __restrict__ qf, const float* __restrict__ kf,
                const float* __restrict__ vf, const float* __restrict__ v1,
                const float* __restrict__ lambp,
                uint16_t* __restrict__ qb, uint16_t* __restrict__ kb,
                uint16_t* __restrict__ vtb) {
  const int t = blockIdx.x, h = blockIdx.y, d = threadIdx.x;
  const size_t idx = (size_t)t * DIM + h * HD + d;
  __shared__ float sred[128];
  __shared__ float svec[128];

  const float lam = lambp[0];
  // v mix + per-head transpose: vtb[h][d][t]
  float vm = (1.0f - lam) * vf[idx] + lam * v1[idx];
  vtb[((size_t)(h * HD + d)) * TSEQ + t] = f2bf(vm);

  const float tt = (float)t;
#pragma unroll
  for (int which = 0; which < 2; ++which) {
    float x = which == 0 ? qf[idx] : kf[idx];
    sred[d] = x * x;
    __syncthreads();
#pragma unroll
    for (int s = 64; s > 0; s >>= 1) {
      if (d < s) sred[d] += sred[d + s];
      __syncthreads();
    }
    float rms = rsqrtf(sred[0] * (1.0f / HD) + EPS);
    __syncthreads();                 // everyone done with sred[0]
    svec[d] = x * rms;
    __syncthreads();
    float out;
    if (d < 64) {
      float ang = tt * __expf(-(float)d * LN1E4_OVER_64);
      out = svec[d] * __cosf(ang) + svec[d + 64] * __sinf(ang);
    } else {
      int j = d - 64;
      float ang = tt * __expf(-(float)j * LN1E4_OVER_64);
      out = -svec[j] * __sinf(ang) + svec[j + 64] * __cosf(ang);
    }
    (which == 0 ? qb : kb)[idx] = f2bf(out);
    __syncthreads();                 // before svec/sred reuse
  }
}

// ---------------- flash attention ----------------
// grid (TSEQ/128, NH), block 256 (8 waves). Each wave: 16 query rows, online
// softmax over 32-key blocks; S via 8 WMMA (K=128), PV via 8 WMMA (O 16x128).
__global__ __launch_bounds__(256)
void k_attention(const uint16_t* __restrict__ qmat,   // (t, h, d) bf16
                 const uint16_t* __restrict__ kmat,   // (t, h, d) bf16
                 const uint16_t* __restrict__ vtb,    // [h][d][t] bf16
                 uint16_t* __restrict__ ymat) {       // (t, h, d) bf16
  __shared__ uint16_t P_lds[8][16][40];               // per-wave P transpose buffer

  const int lane   = threadIdx.x & 31;
  const int w      = threadIdx.x >> 5;
  const int lanelo = lane & 15;
  const int half   = lane >> 4;
  const int h      = blockIdx.y;
  const int qb     = blockIdx.x * 128 + w * 16;       // query row base for this wave
  const float NEG_INF = -__builtin_inff();

  // Q fragments: 4 frags cover head-dim K = 128
  v16bf qfr[4];
  {
    const uint16_t* qrow = qmat + (size_t)(qb + lanelo) * DIM + h * HD;
#pragma unroll
    for (int kk = 0; kk < 4; ++kk)
      qfr[kk] = make_frag2(qrow + 32 * kk + half * 8, qrow + 32 * kk + 16 + half * 8);
  }

  v8f Oacc[8];
#pragma unroll
  for (int i = 0; i < 8; ++i) Oacc[i] = (v8f)(0.0f);
  float mrun[8], lrun[8];
#pragma unroll
  for (int r = 0; r < 8; ++r) { mrun[r] = NEG_INF; lrun[r] = 0.0f; }

  const int nblocks = (qb + 15) / 32 + 1;             // key0 <= qb+15
  for (int blk = 0; blk < nblocks; ++blk) {
    const int key0 = blk * 32;

    // hide first-touch latency of the next K block (L2-resident working set)
    if (blk + 1 < nblocks)
      __builtin_prefetch(kmat + (size_t)(key0 + 32 + lanelo) * DIM + h * HD, 0, 0);

    // S = Q * K^T for 16 rows x 32 keys
    v8f sacc[2];
    sacc[0] = (v8f)(0.0f); sacc[1] = (v8f)(0.0f);
#pragma unroll
    for (int kk = 0; kk < 4; ++kk) {
#pragma unroll
      for (int nf = 0; nf < 2; ++nf) {
        const uint16_t* kp = kmat + (size_t)(key0 + nf * 16 + lanelo) * DIM
                                  + h * HD + 32 * kk + half * 8;
        sacc[nf] = wmma_bf16(qfr[kk], make_frag16(kp), sacc[nf]);
      }
    }

    // causal mask + online softmax (row r lives in VGPR r; col = lane&15)
#pragma unroll
    for (int r = 0; r < 8; ++r) {
      const int qrow = qb + half * 8 + r;
      float s0 = sacc[0][r] * ATT_SCALE;
      float s1 = sacc[1][r] * ATT_SCALE;
      if (key0 + lanelo      > qrow) s0 = NEG_INF;
      if (key0 + 16 + lanelo > qrow) s1 = NEG_INF;

      float rm = fmaxf(s0, s1);
#pragma unroll
      for (int msk = 8; msk >= 1; msk >>= 1)
        rm = fmaxf(rm, __shfl_xor(rm, msk, 32));      // stays within each 16-lane half

      float mn   = fmaxf(mrun[r], rm);
      float corr = __expf(mrun[r] - mn);
      float p0   = __expf(s0 - mn);
      float p1   = __expf(s1 - mn);
      float rs   = p0 + p1;
#pragma unroll
      for (int msk = 8; msk >= 1; msk >>= 1)
        rs += __shfl_xor(rs, msk, 32);

      lrun[r] = lrun[r] * corr + rs;
      mrun[r] = mn;
#pragma unroll
      for (int d = 0; d < 8; ++d) Oacc[d][r] *= corr;

      // stash P (bf16) for transpose into A-fragment layout
      P_lds[w][half * 8 + r][lanelo]      = f2bf(p0);
      P_lds[w][half * 8 + r][16 + lanelo] = f2bf(p1);
    }

    asm volatile("s_wait_dscnt 0" ::: "memory");      // same-wave DS RAW fence

    // P A-fragment (16x32) from LDS
    const uint16_t* prow = &P_lds[w][lanelo][0];
    v16bf pa = make_frag2(prow + half * 8, prow + 16 + half * 8);

    // O += P * V  (V block is 32 keys x 128 dims, read from [h][d][t] layout)
#pragma unroll
    for (int nf2 = 0; nf2 < 8; ++nf2) {
      const uint16_t* vp = vtb + ((size_t)(h * HD + 16 * nf2 + lanelo)) * TSEQ
                               + key0 + half * 8;
      Oacc[nf2] = wmma_bf16(pa, make_frag16(vp), Oacc[nf2]);
    }
    asm volatile("s_wait_dscnt 0" ::: "memory");      // LDS reads done before next overwrite
  }

  // epilogue: scale by 1/l and emit bf16
  float invl[8];
#pragma unroll
  for (int r = 0; r < 8; ++r) invl[r] = 1.0f / lrun[r];
#pragma unroll
  for (int nf2 = 0; nf2 < 8; ++nf2)
#pragma unroll
    for (int r = 0; r < 8; ++r) {
      int row = qb + half * 8 + r;
      ymat[(size_t)row * DIM + h * HD + 16 * nf2 + lanelo] = f2bf(Oacc[nf2][r] * invl[r]);
    }
}

// ---------------- launch ----------------
extern "C" void kernel_launch(void* const* d_in, const int* in_sizes, int n_in,
                              void* d_out, int out_size, void* d_ws, size_t ws_size,
                              hipStream_t stream) {
  const float* x    = (const float*)d_in[0];
  const float* v1   = (const float*)d_in[1];
  const float* Wq   = (const float*)d_in[2];
  const float* Wk   = (const float*)d_in[3];
  const float* Wv   = (const float*)d_in[4];
  const float* Wp   = (const float*)d_in[5];
  const float* lamb = (const float*)d_in[6];
  float* out = (float*)d_out;

  const size_t NE = (size_t)TSEQ * DIM;               // 4M elements
  char* ws = (char*)d_ws;
  size_t off = 0;
  auto carve = [&](size_t bytes) { char* p = ws + off; off += bytes; return p; };
  uint16_t* xb  = (uint16_t*)carve(NE * 2);
  uint16_t* wqb = (uint16_t*)carve(NE * 2);
  uint16_t* wkb = (uint16_t*)carve(NE * 2);
  uint16_t* wvb = (uint16_t*)carve(NE * 2);
  uint16_t* wpb = (uint16_t*)carve(NE * 2);
  float*    qf  = (float*)carve(NE * 4);
  float*    kf  = (float*)carve(NE * 4);
  float*    vf  = (float*)carve(NE * 4);
  uint16_t* qbm = (uint16_t*)carve(NE * 2);
  uint16_t* kbm = (uint16_t*)carve(NE * 2);
  uint16_t* vtb = (uint16_t*)carve(NE * 2);
  uint16_t* yb  = (uint16_t*)carve(NE * 2);
  (void)ws_size; (void)in_sizes; (void)n_in; (void)out_size;

  const int cgrid = (int)((NE + 255) / 256);
  k_f32_to_bf16<<<cgrid, 256, 0, stream>>>(x,  xb,  (int)NE);
  k_f32_to_bf16<<<cgrid, 256, 0, stream>>>(Wq, wqb, (int)NE);
  k_f32_to_bf16<<<cgrid, 256, 0, stream>>>(Wk, wkb, (int)NE);
  k_f32_to_bf16<<<cgrid, 256, 0, stream>>>(Wv, wvb, (int)NE);
  k_f32_to_bf16<<<cgrid, 256, 0, stream>>>(Wp, wpb, (int)NE);

  dim3 gg(DIM / 128, TSEQ / 128);
  k_gemm_bf16<<<gg, 256, 0, stream>>>(xb, wqb, qf, TSEQ, DIM, DIM);
  k_gemm_bf16<<<gg, 256, 0, stream>>>(xb, wkb, kf, TSEQ, DIM, DIM);
  k_gemm_bf16<<<gg, 256, 0, stream>>>(xb, wvb, vf, TSEQ, DIM, DIM);

  dim3 gp(TSEQ, NH);
  k_qkv_post<<<gp, 128, 0, stream>>>(qf, kf, vf, v1, lamb, qbm, kbm, vtb);

  dim3 ga(TSEQ / 128, NH);
  k_attention<<<ga, 256, 0, stream>>>(qbm, kbm, vtb, yb);

  k_gemm_bf16<<<gg, 256, 0, stream>>>(yb, wpb, out, TSEQ, DIM, DIM);
  k_copy_f32<<<cgrid, 256, 0, stream>>>(v1, out + NE, (int)NE);
}
